// AttentionFreeTransformer_84241488544513
// MI455X (gfx1250) — matrix operand order repaired and checked by reading, not compiled, not run on hardware
//
#include <hip/hip_runtime.h>
#include <hip/hip_bf16.h>

#define DIM   1024
#define SEQ   2048
#define HEADS 8
#define HDIM  128
#define NB    4
#define MTOT  (NB * SEQ)   // 8192 rows of the big GEMMs

typedef __attribute__((ext_vector_type(16))) __bf16          v16bf;
typedef __attribute__((ext_vector_type(8)))  float           v8f;
typedef __attribute__((ext_vector_type(4)))  float           f32x4;
typedef __attribute__((ext_vector_type(4)))  unsigned short  u16x4;
typedef __attribute__((ext_vector_type(8)))  unsigned short  u16x8;
typedef __attribute__((ext_vector_type(16))) unsigned short  u16x16;

__device__ __forceinline__ unsigned short f32_to_bf16(float f) {
    unsigned int u = __builtin_bit_cast(unsigned int, f);
    u += 0x7FFFu + ((u >> 16) & 1u);           // round-to-nearest-even
    return (unsigned short)(u >> 16);
}
__device__ __forceinline__ float bf16_to_f32(unsigned short b) {
    return __builtin_bit_cast(float, ((unsigned int)b) << 16);
}
__device__ __forceinline__ v16bf as_bf16(u16x16 u) {
    return __builtin_bit_cast(v16bf, u);
}
__device__ __forceinline__ u16x16 cat8(u16x8 lo, u16x8 hi) {
    return __builtin_shufflevector(lo, hi, 0,1,2,3,4,5,6,7,8,9,10,11,12,13,14,15);
}

// ---------------------------------------------------------------------------
// Pack fp32 -> bf16 (row-major, identical layout)
// ---------------------------------------------------------------------------
__global__ __launch_bounds__(256)
void pack_bf16_kernel(const float* __restrict__ in, unsigned short* __restrict__ out, int count) {
    int i = blockIdx.x * 256 + threadIdx.x;
    if (i < count) out[i] = f32_to_bf16(in[i]);
}

// ---------------------------------------------------------------------------
// Pack fp32 W[in][out] -> bf16 WT[out][in]  (B operand wants K contiguous)
// ---------------------------------------------------------------------------
__global__ __launch_bounds__(256)
void transpose_bf16_kernel(const float* __restrict__ in, unsigned short* __restrict__ out,
                           int rows /*K*/, int cols /*N*/) {
    int idx = blockIdx.x * 256 + threadIdx.x;
    if (idx >= rows * cols) return;
    int o = idx / rows;          // output column (N)
    int i = idx - o * rows;      // input row (K)
    out[idx] = f32_to_bf16(in[(size_t)i * cols + o]);
}

// ---------------------------------------------------------------------------
// Precompute EW = tril(exp(w_aft)) as bf16 [H][T][S].  Vectorized x4.
// One pass: 134 MB read + 67 MB write, and exp computed exactly once per
// element instead of 16x inside the AFT hot loop.
// ---------------------------------------------------------------------------
__global__ __launch_bounds__(256)
void ew_pack_kernel(const float* __restrict__ w, unsigned short* __restrict__ ewb) {
    const int i4 = (blockIdx.x * 256 + threadIdx.x) * 4;   // SEQ % 4 == 0: same row t
    const int t  = (i4 / SEQ) & (SEQ - 1);
    const int s0 = i4 & (SEQ - 1);
    f32x4 v = *(const f32x4*)(w + i4);
    u16x4 o;
    #pragma unroll
    for (int j = 0; j < 4; ++j)
        o[j] = f32_to_bf16((s0 + j <= t) ? __expf(v[j]) : 0.f);
    *(u16x4*)(ewb + i4) = o;
}

// ---------------------------------------------------------------------------
// Unified bf16 WMMA GEMM: C = A[M,K] * B[K,N] (+bias), B given transposed BT[N][K].
// Block tile 128x128, 4 waves (2x2), wave tile 64x64, K step 32.
// B tile staged via LDS (pad 40 shorts/row to spread banks).
// mode 0: outb[((n*H+h)*HDIM+d)*SEQ+s] = bf16(exp(acc+bias))            (ek, head-T layout)
// mode 1: outb[...same...]             = bf16(ek * (acc+bias))          (ekv; ek read from ekin)
// mode 2: outf[row*DIM+col]            = acc + bias                      (final output)
// ---------------------------------------------------------------------------
__global__ __launch_bounds__(128)
void gemm_bf16_kernel(const unsigned short* __restrict__ A, int lda,
                      const unsigned short* __restrict__ BT, int ldb,
                      const float* __restrict__ bias, int K, int mode,
                      unsigned short* __restrict__ outb,
                      const unsigned short* __restrict__ ekin,
                      float* __restrict__ outf) {
    __shared__ unsigned short ldsB[128 * 40];

    const int lane  = threadIdx.x & 31;
    const int wave  = threadIdx.x >> 5;
    const int wrow  = wave >> 1;
    const int wcol  = wave & 1;
    const int bm    = blockIdx.y * 128;
    const int bn    = blockIdx.x * 128;
    const int m0    = bm + wrow * 64;
    const int row   = lane & 15;
    const int khalf = (lane >> 4) & 1;     // A: which K-half this lane holds
    const int kbase = (lane >> 4) << 3;    // B: lanes 16-31 hold K=8..15,24..31

    v8f acc[4][4];
    const v8f vzero = {0.f, 0.f, 0.f, 0.f, 0.f, 0.f, 0.f, 0.f};
    for (int i = 0; i < 4; ++i)
        for (int j = 0; j < 4; ++j) acc[i][j] = vzero;

    for (int k0 = 0; k0 < K; k0 += 32) {
        // Cooperative B tile load: 128 cols x 32 K -> LDS
        {
            const unsigned short* src = BT + (size_t)(bn + threadIdx.x) * ldb + k0;
            unsigned short* dst = ldsB + threadIdx.x * 40;
            *(u16x8*)(dst + 0)  = *(const u16x8*)(src + 0);
            *(u16x8*)(dst + 8)  = *(const u16x8*)(src + 8);
            *(u16x8*)(dst + 16) = *(const u16x8*)(src + 16);
            *(u16x8*)(dst + 24) = *(const u16x8*)(src + 24);
        }
        __syncthreads();

        // A fragments: 4 row-blocks of 16, direct from global (bf16 row-major)
        u16x16 afr[4];
        #pragma unroll
        for (int mi = 0; mi < 4; ++mi) {
            const unsigned short* ap = A + (size_t)(m0 + mi * 16 + row) * lda + k0 + khalf * 8;
            u16x8 lo = *(const u16x8*)ap;
            u16x8 hi = *(const u16x8*)(ap + 16);
            afr[mi] = cat8(lo, hi);
        }

        #pragma unroll
        for (int ni = 0; ni < 4; ++ni) {
            const unsigned short* bp = ldsB + (wcol * 64 + ni * 16 + row) * 40 + kbase;
            u16x8 blo = *(const u16x8*)bp;
            u16x8 bhi = *(const u16x8*)(bp + 16);
            v16bf bfr = as_bf16(cat8(blo, bhi));
            #pragma unroll
            for (int mi = 0; mi < 4; ++mi)
                acc[mi][ni] = __builtin_amdgcn_wmma_f32_16x16x32_bf16(
                    false, as_bf16(afr[mi]), false, bfr, (short)0, acc[mi][ni], false, false);
        }
        __syncthreads();
    }

    // Epilogue. C/D layout: VGPR r -> row = r + 8*(lane>=16), col = lane&15.
    const int rbase = (lane >> 4) << 3;
    #pragma unroll
    for (int mi = 0; mi < 4; ++mi) {
        #pragma unroll
        for (int ni = 0; ni < 4; ++ni) {
            int gc = bn + wcol * 64 + ni * 16 + (lane & 15);
            float bcol = bias[gc];
            v8f c = acc[mi][ni];
            #pragma unroll
            for (int r = 0; r < 8; ++r) {
                int gm = m0 + mi * 16 + rbase + r;
                float val = c[r] + bcol;
                if (mode == 2) {
                    outf[(size_t)gm * DIM + gc] = val;
                } else {
                    int n = gm >> 11, s = gm & (SEQ - 1);
                    int h = gc >> 7, d = gc & (HDIM - 1);
                    size_t o = ((size_t)((n * HEADS + h) * HDIM + d)) * SEQ + s;
                    if (mode == 0) {
                        outb[o] = f32_to_bf16(__expf(val));
                    } else {
                        float e = bf16_to_f32(ekin[o]);
                        outb[o] = f32_to_bf16(e * val);
                    }
                }
            }
        }
    }
}

// ---------------------------------------------------------------------------
// AFT core: per (n,h): Num[T,D] = EW[T,S] @ ekv[S,D], Den[T,D] = EW @ ek.
// PRE=true : EW bf16 precomputed/masked -> A frag = two b128 loads (hot path).
// PRE=false: EW built on the fly from fp32 w_aft (exp + causal mask) --
//            fallback when workspace is too small for the 67 MB EW buffer.
// 4 waves (2x2), wave tile 64 rows x 32 d-cols, num+den fused (16 WMMA/Kstep).
// Causal: K loop stops at the diagonal block of this row tile.
// Output: aft bf16 row-major [N*S][DIM] for the final projection.
// ---------------------------------------------------------------------------
template <bool PRE>
__global__ __launch_bounds__(128)
void aft_core_kernel(const float* __restrict__ w,             // [H][S][S] fp32
                     const unsigned short* __restrict__ ewb,  // [H][S][S] bf16 (PRE)
                     const unsigned short* __restrict__ ekvT, // [N][H][HDIM][S]
                     const unsigned short* __restrict__ ekT,  // [N][H][HDIM][S]
                     unsigned short* __restrict__ aftb) {     // [N*S][DIM]
    const int lane  = threadIdx.x & 31;
    const int wave  = threadIdx.x >> 5;
    const int wrow  = wave >> 1;
    const int wcol  = wave & 1;
    const int nh    = blockIdx.z;
    const int n     = nh >> 3;
    const int h     = nh & 7;
    const int t0    = blockIdx.y * 128 + wrow * 64;
    const int d0    = blockIdx.x * 64 + wcol * 32;
    const int row   = lane & 15;
    const int khalf = (lane >> 4) & 1;
    const int kbase = (lane >> 4) << 3;

    const float*          wb  = w   + (size_t)h * SEQ * SEQ;
    const unsigned short* eb  = ewb + (size_t)h * SEQ * SEQ;
    const unsigned short* bv  = ekvT + (size_t)(n * HEADS + h) * HDIM * SEQ;
    const unsigned short* be  = ekT  + (size_t)(n * HEADS + h) * HDIM * SEQ;

    v8f num[4][2], den[4][2];
    const v8f vzero = {0.f, 0.f, 0.f, 0.f, 0.f, 0.f, 0.f, 0.f};
    for (int i = 0; i < 4; ++i)
        for (int j = 0; j < 2; ++j) { num[i][j] = vzero; den[i][j] = vzero; }

    const int kmax = blockIdx.y * 128 + 128;   // causal: need s <= max t of this tile
    for (int k0 = 0; k0 < kmax; k0 += 32) {
        // A fragments (EW), masked at source
        u16x16 afr[4];
        #pragma unroll
        for (int mi = 0; mi < 4; ++mi) {
            const int t = t0 + mi * 16 + row;
            if constexpr (PRE) {
                const unsigned short* ap = eb + (size_t)t * SEQ + k0 + khalf * 8;
                afr[mi] = cat8(*(const u16x8*)ap, *(const u16x8*)(ap + 16));
            } else {
                const float* ap = wb + (size_t)t * SEQ + k0 + khalf * 8;
                f32x4 w0 = *(const f32x4*)(ap + 0);
                f32x4 w1 = *(const f32x4*)(ap + 4);
                f32x4 w2 = *(const f32x4*)(ap + 16);
                f32x4 w3 = *(const f32x4*)(ap + 20);
                u16x16 f;
                #pragma unroll
                for (int j = 0; j < 4; ++j) {
                    int s = k0 + khalf * 8 + j;
                    f[j]      = f32_to_bf16((s      <= t) ? __expf(w0[j]) : 0.f);
                    f[4 + j]  = f32_to_bf16((s + 4  <= t) ? __expf(w1[j]) : 0.f);
                    f[8 + j]  = f32_to_bf16((s + 16 <= t) ? __expf(w2[j]) : 0.f);
                    f[12 + j] = f32_to_bf16((s + 20 <= t) ? __expf(w3[j]) : 0.f);
                }
                afr[mi] = f;
            }
        }

        #pragma unroll
        for (int ni = 0; ni < 2; ++ni) {
            const size_t boff = (size_t)(d0 + ni * 16 + (lane & 15)) * SEQ + k0 + kbase;
            v16bf bfv = as_bf16(cat8(*(const u16x8*)(bv + boff), *(const u16x8*)(bv + boff + 16)));
            v16bf bfe = as_bf16(cat8(*(const u16x8*)(be + boff), *(const u16x8*)(be + boff + 16)));
            #pragma unroll
            for (int mi = 0; mi < 4; ++mi) {
                num[mi][ni] = __builtin_amdgcn_wmma_f32_16x16x32_bf16(
                    false, as_bf16(afr[mi]), false, bfv, (short)0, num[mi][ni], false, false);
                den[mi][ni] = __builtin_amdgcn_wmma_f32_16x16x32_bf16(
                    false, as_bf16(afr[mi]), false, bfe, (short)0, den[mi][ni], false, false);
            }
        }
    }

    const int rbase = (lane >> 4) << 3;
    #pragma unroll
    for (int mi = 0; mi < 4; ++mi) {
        #pragma unroll
        for (int ni = 0; ni < 2; ++ni) {
            int d = d0 + ni * 16 + (lane & 15);
            #pragma unroll
            for (int r = 0; r < 8; ++r) {
                int t = t0 + mi * 16 + rbase + r;
                float a = num[mi][ni][r] / den[mi][ni][r];   // den > 0 always (exp sums)
                aftb[(size_t)(n * SEQ + t) * DIM + h * HDIM + d] = f32_to_bf16(a);
            }
        }
    }
}

// ---------------------------------------------------------------------------
// Host-side launch
// ---------------------------------------------------------------------------
extern "C" void kernel_launch(void* const* d_in, const int* in_sizes, int n_in,
                              void* d_out, int out_size, void* d_ws, size_t ws_size,
                              hipStream_t stream) {
    const float* x     = (const float*)d_in[0];
    const float* Wk    = (const float*)d_in[1];
    const float* bk    = (const float*)d_in[2];
    const float* Wv    = (const float*)d_in[3];
    const float* bv    = (const float*)d_in[4];
    const float* w_aft = (const float*)d_in[5];
    const float* Wo    = (const float*)d_in[6];
    const float* bo    = (const float*)d_in[7];
    float* out = (float*)d_out;

    // Workspace carve-up (bf16 element counts)
    const size_t n_xb  = (size_t)MTOT * DIM;
    const size_t n_w   = (size_t)DIM * DIM;
    const size_t n_ekt = (size_t)NB * HEADS * HDIM * SEQ;
    const size_t n_aft = (size_t)MTOT * DIM;
    const size_t n_ew  = (size_t)HEADS * SEQ * SEQ;
    const size_t base_elems = n_xb + 3 * n_w + 2 * n_ekt + n_aft;      // ~73 MB
    const bool   use_pre    = ws_size >= (base_elems + n_ew) * 2;      // +67 MB for EW

    unsigned short* ws = (unsigned short*)d_ws;
    unsigned short* xb   = ws;  ws += n_xb;
    unsigned short* WkT  = ws;  ws += n_w;
    unsigned short* WvT  = ws;  ws += n_w;
    unsigned short* WoT  = ws;  ws += n_w;
    unsigned short* ekT  = ws;  ws += n_ekt;   // exp(k), [n][h][d][s]
    unsigned short* ekvT = ws;  ws += n_ekt;   // exp(k)*v
    unsigned short* aftb = ws;  ws += n_aft;   // aft bf16 row-major
    unsigned short* ewb  = ws;                 // tril(exp(w_aft)) bf16 (optional)

    // Stage 0: packs
    pack_bf16_kernel<<<(MTOT * DIM + 255) / 256, 256, 0, stream>>>(x, xb, MTOT * DIM);
    transpose_bf16_kernel<<<(DIM * DIM + 255) / 256, 256, 0, stream>>>(Wk, WkT, DIM, DIM);
    transpose_bf16_kernel<<<(DIM * DIM + 255) / 256, 256, 0, stream>>>(Wv, WvT, DIM, DIM);
    transpose_bf16_kernel<<<(DIM * DIM + 255) / 256, 256, 0, stream>>>(Wo, WoT, DIM, DIM);
    if (use_pre)
        ew_pack_kernel<<<(int)(n_ew / 4 / 256), 256, 0, stream>>>(w_aft, ewb);

    // Stage 1: k projection -> ek ; v projection -> ekv (reads ek)
    dim3 gproj(DIM / 128, MTOT / 128, 1);      // (8, 64)
    gemm_bf16_kernel<<<gproj, 128, 0, stream>>>(xb, DIM, WkT, DIM, bk, DIM, 0, ekT, nullptr, nullptr);
    gemm_bf16_kernel<<<gproj, 128, 0, stream>>>(xb, DIM, WvT, DIM, bv, DIM, 1, ekvT, ekT, nullptr);

    // Stage 2: AFT einsum + normalization
    dim3 gaft(HDIM / 64, SEQ / 128, NB * HEADS);   // (2, 16, 32)
    if (use_pre)
        aft_core_kernel<true><<<gaft, 128, 0, stream>>>(w_aft, ewb, ekvT, ekT, aftb);
    else
        aft_core_kernel<false><<<gaft, 128, 0, stream>>>(w_aft, nullptr, ekvT, ekT, aftb);

    // Stage 3: output projection (fp32 out)
    gemm_bf16_kernel<<<gproj, 128, 0, stream>>>(aftb, DIM, WoT, DIM, bo, DIM, 2, nullptr, nullptr, out);
}
